// FastSpeech2_29429115912694
// MI455X (gfx1250) — matrix operand order repaired and checked by reading, not compiled, not run on hardware
//
#include <hip/hip_runtime.h>
#include <hip/hip_bf16.h>

#define DEV __device__ __forceinline__

constexpr int kB = 8, kTT = 256, kTM = 2048;
constexpr int kD = 256, kH = 2, kDK = 128, kDFF = 1024, kL = 4;
constexpr int kDPF = 256, kNM = 80;
constexpr int kHD = kH * kDK; // 256
constexpr long kMEL_TOTAL = (long)kB * kTM * kNM; // 1,310,720

typedef __attribute__((ext_vector_type(16))) __bf16 v16bf;
typedef __attribute__((ext_vector_type(8)))  float  v8f;

union AFrag { v16bf v; unsigned int u[8]; };
union CFrag { v8f  v; float        f[8]; };

DEV unsigned short f2bf(float x) {
  unsigned int u = __float_as_uint(x);
  u += 0x7FFFu + ((u >> 16) & 1u);       // round-to-nearest-even
  return (unsigned short)(u >> 16);
}
DEV float bf2f(unsigned short b) { return __uint_as_float(((unsigned int)b) << 16); }

// A-fragment (16x32 bf16) K index for vgpr v, lane-group hi (ISA 7.12.2)
DEV int kofA(int v, int hi) { return (v < 4) ? (8 * hi + 2 * v) : (16 + 8 * hi + 2 * (v - 4)); }

DEV float redmax16(float x) {
  x = fmaxf(x, __shfl_xor(x, 1));
  x = fmaxf(x, __shfl_xor(x, 2));
  x = fmaxf(x, __shfl_xor(x, 4));
  x = fmaxf(x, __shfl_xor(x, 8));
  return x;
}
DEV float redsum16(float x) {
  x += __shfl_xor(x, 1); x += __shfl_xor(x, 2);
  x += __shfl_xor(x, 4); x += __shfl_xor(x, 8);
  return x;
}
DEV float redsum32(float x) {
  x += __shfl_xor(x, 1);  x += __shfl_xor(x, 2);
  x += __shfl_xor(x, 4);  x += __shfl_xor(x, 8);
  x += __shfl_xor(x, 16);
  return x;
}

DEV v8f wmma_bf(const AFrag& a, const AFrag& b, const v8f& c) {
  return __builtin_amdgcn_wmma_f32_16x16x32_bf16(false, a.v, false, b.v, (short)0, c,
                                                 false, false);
}

// ---------------------------------------------------------------------------
// Generic multi-tap bf16 WMMA GEMM:
//   C[m][n] = sum_tap sum_k A[(b,t+tap-pad)][k] * Bt[tap][n][k]  (+bias, relu)
// Rows m = b*T + t; out-of-range time taps read zero (conv padding).
// Each wave computes a 32x64 output tile (2 M-subtiles x 4 N-subtiles), so
// each B fragment feeds two WMMAs and B traffic is amortized across M.
// ---------------------------------------------------------------------------
__global__ __launch_bounds__(256) void gemm_bf16(
    const unsigned short* __restrict__ A, const unsigned short* __restrict__ Bt,
    const float* __restrict__ bias, float* __restrict__ Cf,
    unsigned short* __restrict__ Cbf,
    int M, int N, int K, int T, int taps, int pad, int relu)
{
  const int lane = threadIdx.x & 31;
  const int wid  = threadIdx.x >> 5;
  const int r    = lane & 15;
  const int hi   = lane >> 4;

  const int  mTiles  = M >> 5;                 // 32-row macro tiles
  const int  nGroups = (N + 63) >> 6;
  const long w       = (long)blockIdx.x * 8 + wid;
  if (w >= (long)mTiles * nGroups) return;
  const int m0 = (int)(w % mTiles) << 5;
  const int n0 = (int)(w / mTiles) << 6;

  CFrag acc[2][4];
#pragma unroll
  for (int g = 0; g < 2; ++g)
#pragma unroll
    for (int t = 0; t < 4; ++t)
#pragma unroll
      for (int j = 0; j < 8; ++j) acc[g][t].f[j] = 0.0f;

  const long tapStride = (long)N * K;
  const int  row0 = m0 + r,      row1 = m0 + 16 + r;
  const int  b0 = row0 / T,      t0 = row0 % T;
  const int  b1 = row1 / T,      t1 = row1 % T;
  // A fragment: one contiguous 16-halfword run per lane at 16*hi, as B (ka0/ka4 unused)
  const int ka0 = kofA(0, hi);                 // table-order start of v=0..3, unused
  const int ka4 = kofA(4, hi);                 // table-order start of v=4..7, unused
  const int kb0 = 16 * hi;                     // B fragment run (16 halfwords)

  for (int tap = 0; tap < taps; ++tap) {
    const int  ts0 = t0 + tap - pad;
    const int  ts1 = t1 + tap - pad;
    const bool valid0 = (ts0 >= 0) && (ts0 < T);
    const bool valid1 = (ts1 >= 0) && (ts1 < T);
    const unsigned short* Ar0 = A + ((long)b0 * T + ts0) * K;
    const unsigned short* Ar1 = A + ((long)b1 * T + ts1) * K;
    const unsigned short* Bb  = Bt + (long)tap * tapStride + (long)r * K; // col base

    for (int k0 = 0; k0 < K; k0 += 32) {
      if (k0 + 32 < K) {                       // speculative prefetch of next K-step
        __builtin_prefetch(Ar0 + k0 + 32, 0, 3);
        __builtin_prefetch(Ar1 + k0 + 32, 0, 3);
      }
      AFrag a0, a1;
#pragma unroll
      for (int v = 0; v < 8; ++v) {
        const int kk = k0 + 16 * hi + 2 * v;
        a0.u[v] = valid0 ? *(const unsigned int*)(Ar0 + kk) : 0u;
        a1.u[v] = valid1 ? *(const unsigned int*)(Ar1 + kk) : 0u;
      }
#pragma unroll
      for (int nt = 0; nt < 4; ++nt) {
        const int nb = n0 + nt * 16;
        if (nb >= N) break;                    // wave-uniform
        AFrag bf;
        const unsigned short* Bp = Bb + (long)nb * K + k0 + kb0;
#pragma unroll
        for (int v = 0; v < 8; ++v)
          bf.u[v] = *(const unsigned int*)(Bp + 2 * v);
        acc[0][nt].v = wmma_bf(a0, bf, acc[0][nt].v);
        acc[1][nt].v = wmma_bf(a1, bf, acc[1][nt].v);
      }
    }
  }

#pragma unroll
  for (int nt = 0; nt < 4; ++nt) {
    const int nb = n0 + nt * 16;
    if (nb >= N) break;
    const int  col = nb + r;
    const float bv = bias ? bias[col] : 0.0f;
#pragma unroll
    for (int g = 0; g < 2; ++g)
#pragma unroll
      for (int v = 0; v < 8; ++v) {
        const int rw = m0 + g * 16 + v + 8 * hi;
        float x = acc[g][nt].f[v] + bv;
        if (relu) x = fmaxf(x, 0.0f);
        if (Cf)  Cf [(long)rw * N + col] = x;
        if (Cbf) Cbf[(long)rw * N + col] = f2bf(x);
      }
  }
}

// ---------------------------------------------------------------------------
// Fused flash attention. One wave = 16 queries x DK=128 for one (b,h).
// Q,K bf16 [b][t][h*DK+dk]; Vt bf16 [b][h][dk][t]; seq int (0 => key masked).
// O bf16 [b][t][h*DK+dk] = softmax(QK^T/sqrt(DK) with mask) V.
// ---------------------------------------------------------------------------
__global__ __launch_bounds__(256) void attn_fused(
    const unsigned short* __restrict__ Q, const unsigned short* __restrict__ Kb,
    const unsigned short* __restrict__ Vt, const int* __restrict__ seq,
    unsigned short* __restrict__ O, int Bn, int T)
{
  __shared__ unsigned short pbuf[8][16 * 32];   // per-wave P staging (bf16)

  const int lane = threadIdx.x & 31;
  const int wid  = threadIdx.x >> 5;
  const int r    = lane & 15;
  const int hi   = lane >> 4;

  const int  qTiles = T >> 4;
  const long w      = (long)blockIdx.x * 8 + wid;
  if (w >= (long)Bn * kH * qTiles) return;
  const int qt = (int)(w % qTiles);
  const int h  = (int)((w / qTiles) % kH);
  const int b  = (int)(w / ((long)qTiles * kH));
  const int q0 = qt << 4;

  // Preload Q fragments (4 k-steps covering DK=128)
  AFrag qf[4];
  const unsigned short* Qrow = Q + ((long)(b * T + q0 + r) * kHD + h * kDK);
#pragma unroll
  for (int ks = 0; ks < 4; ++ks)
#pragma unroll
    for (int v = 0; v < 8; ++v)
      qf[ks].u[v] = *(const unsigned int*)(Qrow + ks * 32 + 16 * hi + 2 * v);

  CFrag acc[8];
#pragma unroll
  for (int d = 0; d < 8; ++d)
#pragma unroll
    for (int v = 0; v < 8; ++v) acc[d].f[v] = 0.0f;
  float m[8], l[8];
#pragma unroll
  for (int v = 0; v < 8; ++v) { m[v] = -3.0e38f; l[v] = 0.0f; }

  const float scale = 0.08838834764831845f;   // 1/sqrt(128)
  const unsigned short* Vbase = Vt + ((long)(b * kH + h) * kDK) * T;
  const unsigned int*   pb32  = (const unsigned int*)&pbuf[wid][0];

  for (int key0 = 0; key0 < T; key0 += 32) {
    CFrag s[2];
#pragma unroll
    for (int j = 0; j < 2; ++j) {
#pragma unroll
      for (int v = 0; v < 8; ++v) s[j].f[v] = 0.0f;
      const unsigned short* Krow =
          Kb + ((long)(b * T + key0 + j * 16 + r) * kHD + h * kDK);
#pragma unroll
      for (int ks = 0; ks < 4; ++ks) {
        AFrag kf;
#pragma unroll
        for (int v = 0; v < 8; ++v)
          kf.u[v] = *(const unsigned int*)(Krow + ks * 32 + 16 * hi + 2 * v);
        s[j].v = wmma_bf(qf[ks], kf, s[j].v);
      }
      // scale + key mask (per-column)
      const bool msk = (seq[(long)b * T + key0 + j * 16 + r] == 0);
#pragma unroll
      for (int v = 0; v < 8; ++v) {
        float x = s[j].f[v] * scale;
        s[j].f[v] = msk ? -1.0e9f : x;
      }
    }
    // online softmax (row stats live in 16-lane groups: rows v+8*hi)
    float al[8];
#pragma unroll
    for (int v = 0; v < 8; ++v) {
      const float mch = redmax16(fmaxf(s[0].f[v], s[1].f[v]));
      const float mn  = fmaxf(m[v], mch);
      al[v] = __expf(m[v] - mn);
      m[v]  = mn;
      const float p0 = __expf(s[0].f[v] - mn);
      const float p1 = __expf(s[1].f[v] - mn);
      s[0].f[v] = p0; s[1].f[v] = p1;
      l[v] = l[v] * al[v] + redsum16(p0 + p1);
    }
#pragma unroll
    for (int d = 0; d < 8; ++d)
#pragma unroll
      for (int v = 0; v < 8; ++v) acc[d].f[v] *= al[v];

    // stage P (C-layout) into LDS, reload in A-layout
#pragma unroll
    for (int j = 0; j < 2; ++j)
#pragma unroll
      for (int v = 0; v < 8; ++v)
        pbuf[wid][(v + 8 * hi) * 32 + j * 16 + r] = f2bf(s[j].f[v]);
    asm volatile("s_wait_dscnt 0x0" ::: "memory");
    AFrag pf;
#pragma unroll
    for (int v = 0; v < 8; ++v)
      pf.u[v] = pb32[(r * 32 + 16 * hi + 2 * v) >> 1];

    // O += P @ V  (8 DK-subtiles)
#pragma unroll
    for (int d = 0; d < 8; ++d) {
      AFrag vf;
      const unsigned short* Vrow = Vbase + (long)(d * 16 + r) * T + key0;
#pragma unroll
      for (int v = 0; v < 8; ++v)
        vf.u[v] = *(const unsigned int*)(Vrow + 16 * hi + 2 * v);
      acc[d].v = wmma_bf(pf, vf, acc[d].v);
    }
  }

#pragma unroll
  for (int d = 0; d < 8; ++d)
#pragma unroll
    for (int v = 0; v < 8; ++v) {
      const int qrow = q0 + v + 8 * hi;
      O[((long)(b * T + qrow)) * kHD + h * kDK + d * 16 + r] =
          f2bf(acc[d].f[v] / l[v]);
    }
}

// ---------------------------------------------------------------------------
// Residual + LayerNorm (+optional ReLU, +optional non-pad mask). D must be 256.
// One wave per row; dual fp32/bf16 output.
// ---------------------------------------------------------------------------
__global__ __launch_bounds__(256) void ln_kernel(
    const float* __restrict__ X, const float* __restrict__ Res,
    const float* __restrict__ g, const float* __restrict__ bb,
    const int* __restrict__ seq, int relu,
    float* __restrict__ Of, unsigned short* __restrict__ Obf, int Mrows)
{
  const int lane = threadIdx.x & 31;
  const int wid  = threadIdx.x >> 5;
  const long row = (long)blockIdx.x * 8 + wid;
  if (row >= Mrows) return;

  const float* xr = X + row * kD;
  const float* rr = Res ? Res + row * kD : nullptr;
  float vals[8], s = 0.0f, s2 = 0.0f;
#pragma unroll
  for (int i = 0; i < 8; ++i) {
    const int c = lane * 8 + i;
    float x = xr[c] + (rr ? rr[c] : 0.0f);
    vals[i] = x; s += x; s2 += x * x;
  }
  s  = redsum32(s);
  s2 = redsum32(s2);
  const float mean = s * (1.0f / kD);
  const float var  = s2 * (1.0f / kD) - mean * mean;
  const float inv  = rsqrtf(var + 1e-5f);
  const float np   = seq ? ((seq[row] != 0) ? 1.0f : 0.0f) : 1.0f;
#pragma unroll
  for (int i = 0; i < 8; ++i) {
    const int c = lane * 8 + i;
    float y = (vals[i] - mean) * inv * g[c] + bb[c];
    if (relu) y = fmaxf(y, 0.0f);
    y *= np;
    Of[row * kD + c] = y;
    if (Obf) Obf[row * kD + c] = f2bf(y);
  }
}

// ---------------------------------------------------------------------------
__global__ void embed_enc(const int* __restrict__ seq, const int* __restrict__ pos,
                          const float* __restrict__ wemb, const float* __restrict__ pemb,
                          float* __restrict__ xf, unsigned short* __restrict__ xbf)
{
  const long i = (long)blockIdx.x * 256 + threadIdx.x;
  if (i >= (long)kB * kTT * kD) return;
  const long rowi = i / kD;
  const int  c    = (int)(i % kD);
  const float v = wemb[(long)seq[rowi] * kD + c] + pemb[(long)pos[rowi] * kD + c];
  xf[i] = v; xbf[i] = f2bf(v);
}

__global__ void vtrans(const unsigned short* __restrict__ V,
                       unsigned short* __restrict__ Vt, int Bn, int T)
{
  const long i = (long)blockIdx.x * 256 + threadIdx.x;
  if (i >= (long)Bn * T * kHD) return;
  const int dk = (int)(i % kDK);
  const int h  = (int)((i / kDK) % kH);
  const int t  = (int)((i / (long)kHD) % T);
  const int b  = (int)(i / ((long)kHD * T));
  Vt[(((long)b * kH + h) * kDK + dk) * T + t] = V[i];
}

__global__ void cumsum_k(const int* __restrict__ dur, int* __restrict__ cum)
{
  const int b = threadIdx.x;
  if (b >= kB) return;
  int s = 0;
  for (int t = 0; t < kTT; ++t) { s += dur[b * kTT + t]; cum[b * kTT + t] = s; }
}

__global__ __launch_bounds__(256) void lengthreg(
    const float* __restrict__ xe, const int* __restrict__ cum,
    const int* __restrict__ melpos, const float* __restrict__ dpemb,
    float* __restrict__ yf, unsigned short* __restrict__ ybf)
{
  const int lane = threadIdx.x & 31;
  const int wid  = threadIdx.x >> 5;
  const long row = (long)blockIdx.x * 8 + wid;
  if (row >= (long)kB * kTM) return;
  const int b = (int)(row / kTM);
  const int t = (int)(row % kTM);
  const int* c = cum + b * kTT;
  int lo = 0, hiE = kTT;                  // searchsorted(..., 'right')
  while (lo < hiE) { int mid = (lo + hiE) >> 1; if (c[mid] <= t) lo = mid + 1; else hiE = mid; }
  const int  valid = (t < c[kTT - 1]) ? 1 : 0;
  const int  idx   = lo < (kTT - 1) ? lo : (kTT - 1);
  const int  mp    = melpos[row];
#pragma unroll
  for (int i = 0; i < 8; ++i) {
    const int col = lane * 8 + i;
    float v = (valid ? xe[((long)b * kTT + idx) * kD + col] : 0.0f) +
              dpemb[(long)mp * kD + col];
    yf[row * kD + col] = v;
    ybf[row * kD + col] = f2bf(v);
  }
}

__global__ __launch_bounds__(256) void durdot(
    const float* __restrict__ hbuf, const float* __restrict__ lw,
    const float* __restrict__ lb, float* __restrict__ out)
{
  const int lane = threadIdx.x & 31;
  const int wid  = threadIdx.x >> 5;
  const long row = (long)blockIdx.x * 8 + wid;
  if (row >= (long)kB * kTT) return;
  float s = 0.0f;
#pragma unroll
  for (int i = 0; i < 8; ++i) {
    const int c = lane * 8 + i;
    s += hbuf[row * kDPF + c] * lw[c];
  }
  s = redsum32(s);
  if (lane == 0) out[row] = fmaxf(s + lb[0], 0.0f);
}

// --------------------------- weight preparation ----------------------------
__global__ void w_transpose_bf(const float* __restrict__ src,
                               unsigned short* __restrict__ dst, int K, int N)
{
  const long i = (long)blockIdx.x * 256 + threadIdx.x;
  if (i >= (long)K * N) return;
  const int k = (int)(i / N), n = (int)(i % N);
  dst[(long)n * K + k] = f2bf(src[i]);
}
__global__ void w_copy_bf(const float* __restrict__ src,
                          unsigned short* __restrict__ dst, long n)
{
  const long i = (long)blockIdx.x * 256 + threadIdx.x;
  if (i < n) dst[i] = f2bf(src[i]);
}
// src [F][C][taps] -> dst [tap][F][C]
__global__ void w_convprep(const float* __restrict__ src,
                           unsigned short* __restrict__ dst, int F, int C, int taps)
{
  const long i = (long)blockIdx.x * 256 + threadIdx.x;
  if (i >= (long)taps * F * C) return;
  const int tap = (int)(i / ((long)F * C));
  const long rem = i % ((long)F * C);
  const int f = (int)(rem / C), c = (int)(rem % C);
  dst[i] = f2bf(src[((long)f * C + c) * taps + tap]);
}

// ===========================================================================
struct StackIdx { int bk, bo, bq, bv, c1b, c1w, c2b, c2w, ln1b, ln1g, ln2b, ln2g, wk, wo, wq, wv; };

extern "C" void kernel_launch(void* const* d_in, const int* in_sizes, int n_in,
                              void* d_out, int out_size, void* d_ws, size_t ws_size,
                              hipStream_t stream)
{
  (void)in_sizes; (void)n_in; (void)out_size; (void)ws_size;
  // Flat input order: 4 int arrays, then params pytree (dicts sorted by key).
  const int* src_seq = (const int*)d_in[0];
  const int* src_pos = (const int*)d_in[1];
  const int* mel_pos = (const int*)d_in[2];
  const int* len_tgt = (const int*)d_in[3];
  auto F = [&](int i) { return (const float*)d_in[i]; };
  const StackIdx DEC = {4, 5, 6, 7, 8, 9, 10, 11, 12, 13, 14, 15, 16, 17, 18, 19};
  const StackIdx ENC = {31, 32, 33, 34, 35, 36, 37, 38, 39, 40, 41, 42, 43, 44, 45, 46};
  enum { DEC_POS_EMB = 20, DP_C1B = 21, DP_C1W = 22, DP_C2B = 23, DP_C2W = 24,
         DP_LINB = 25, DP_LINW = 26, DP_LN1B = 27, DP_LN1G = 28, DP_LN2B = 29,
         DP_LN2G = 30, ENC_POS_EMB = 47, MEL_B = 48, MEL_W = 49, WORD_EMB = 50 };

  // ------------------- workspace carve-up -------------------
  char* wsb = (char*)d_ws;
  size_t off = 0;
  auto alloc = [&](size_t bytes) -> void* {
    void* p = wsb + off;
    off += (bytes + 255) & ~size_t(255);
    return p;
  };
  struct StackW { unsigned short *wqT, *wkT, *wvT, *woT, *c1T, *c2T; };
  StackW SW[2];
  for (int s = 0; s < 2; ++s) {
    SW[s].wqT = (unsigned short*)alloc((size_t)kL * kD * kD * 2);
    SW[s].wkT = (unsigned short*)alloc((size_t)kL * kD * kD * 2);
    SW[s].wvT = (unsigned short*)alloc((size_t)kL * kD * kD * 2);
    SW[s].woT = (unsigned short*)alloc((size_t)kL * kD * kD * 2);
    SW[s].c1T = (unsigned short*)alloc((size_t)kL * 9 * kDFF * kD * 2);
    SW[s].c2T = (unsigned short*)alloc((size_t)kL * kD * kDFF * 2);
  }
  unsigned short* melT  = (unsigned short*)alloc((size_t)kNM * kD * 2);
  unsigned short* dpc1T = (unsigned short*)alloc((size_t)3 * kDPF * kD * 2);
  unsigned short* dpc2T = (unsigned short*)alloc((size_t)3 * kDPF * kDPF * 2);

  float*          xef  = (float*)alloc((size_t)kB * kTT * kD * 4);
  unsigned short* xebf = (unsigned short*)alloc((size_t)kB * kTT * kD * 2);
  float*          xdf  = (float*)alloc((size_t)kB * kTM * kD * 4);
  unsigned short* xdbf = (unsigned short*)alloc((size_t)kB * kTM * kD * 2);
  unsigned short* qbf  = (unsigned short*)alloc((size_t)kB * kTM * kHD * 2);
  unsigned short* kbf  = (unsigned short*)alloc((size_t)kB * kTM * kHD * 2);
  unsigned short* vbf  = (unsigned short*)alloc((size_t)kB * kTM * kHD * 2);
  unsigned short* vtbf = (unsigned short*)alloc((size_t)kB * kTM * kHD * 2);
  unsigned short* obf  = (unsigned short*)alloc((size_t)kB * kTM * kHD * 2);
  unsigned short* h1bf = (unsigned short*)alloc((size_t)kB * kTM * kDFF * 2);
  float*          tmpf = (float*)alloc((size_t)kB * kTM * kD * 4);
  float*          dpaf = (float*)alloc((size_t)kB * kTT * kDPF * 4);
  unsigned short* dpab = (unsigned short*)alloc((size_t)kB * kTT * kDPF * 2);
  float*          dpbf = (float*)alloc((size_t)kB * kTT * kDPF * 4);
  int*            cum  = (int*)alloc((size_t)kB * kTT * 4);

  float* out = (float*)d_out;                 // mel [B,TMEL,80], then dur [B,TTEXT]

  auto cdiv = [](long a, long b) { return (unsigned)((a + b - 1) / b); };

  // ------------------- weight prep (bf16 / transposed) -------------------
  for (int s = 0; s < 2; ++s) {
    const StackIdx& SI = s ? DEC : ENC;
    for (int i = 0; i < kL; ++i) {
      const long wsz = (long)kD * kD;
      w_transpose_bf<<<cdiv(wsz, 256), 256, 0, stream>>>(F(SI.wq) + i * wsz, SW[s].wqT + i * wsz, kD, kD);
      w_transpose_bf<<<cdiv(wsz, 256), 256, 0, stream>>>(F(SI.wk) + i * wsz, SW[s].wkT + i * wsz, kD, kD);
      w_transpose_bf<<<cdiv(wsz, 256), 256, 0, stream>>>(F(SI.wv) + i * wsz, SW[s].wvT + i * wsz, kD, kD);
      w_transpose_bf<<<cdiv(wsz, 256), 256, 0, stream>>>(F(SI.wo) + i * wsz, SW[s].woT + i * wsz, kD, kD);
      const long c1sz = (long)9 * kDFF * kD;
      w_convprep<<<cdiv(c1sz, 256), 256, 0, stream>>>(F(SI.c1w) + (long)i * kDFF * kD * 9,
                                                      SW[s].c1T + (long)i * c1sz, kDFF, kD, 9);
      const long c2sz = (long)kD * kDFF;      // c2_w [D][DFF][1] is already Bt[N][K]
      w_copy_bf<<<cdiv(c2sz, 256), 256, 0, stream>>>(F(SI.c2w) + (long)i * c2sz,
                                                     SW[s].c2T + (long)i * c2sz, c2sz);
    }
  }
  w_transpose_bf<<<cdiv((long)kD * kNM, 256), 256, 0, stream>>>(F(MEL_W), melT, kD, kNM);
  w_convprep<<<cdiv((long)3 * kDPF * kD, 256), 256, 0, stream>>>(F(DP_C1W), dpc1T, kDPF, kD, 3);
  w_convprep<<<cdiv((long)3 * kDPF * kDPF, 256), 256, 0, stream>>>(F(DP_C2W), dpc2T, kDPF, kDPF, 3);

  auto gemm = [&](const unsigned short* A, const unsigned short* Bt, const float* bias,
                  float* Cf, unsigned short* Cbf, int M, int N, int K, int T,
                  int taps, int pad, int relu) {
    const long waves = (long)(M / 32) * ((N + 63) / 64);
    gemm_bf16<<<cdiv(waves, 8), 256, 0, stream>>>(A, Bt, bias, Cf, Cbf, M, N, K, T, taps, pad, relu);
  };

  auto run_stack = [&](const StackW& W, const StackIdx& SI, float* xf, unsigned short* xbf,
                       int T, const int* seq) {
    const int  Mr  = kB * T;
    const long wsz = (long)kD * kD;
    for (int i = 0; i < kL; ++i) {
      gemm(xbf, W.wqT + i * wsz, F(SI.bq) + i * kD, nullptr, qbf, Mr, kD, kD, T, 1, 0, 0);
      gemm(xbf, W.wkT + i * wsz, F(SI.bk) + i * kD, nullptr, kbf, Mr, kD, kD, T, 1, 0, 0);
      gemm(xbf, W.wvT + i * wsz, F(SI.bv) + i * kD, nullptr, vbf, Mr, kD, kD, T, 1, 0, 0);
      vtrans<<<cdiv((long)Mr * kHD, 256), 256, 0, stream>>>(vbf, vtbf, kB, T);
      const long aw = (long)kB * kH * (T / 16);
      attn_fused<<<cdiv(aw, 8), 256, 0, stream>>>(qbf, kbf, vtbf, seq, obf, kB, T);
      gemm(obf, W.woT + i * wsz, F(SI.bo) + i * kD, tmpf, nullptr, Mr, kD, kD, T, 1, 0, 0);
      ln_kernel<<<cdiv(Mr, 8), 256, 0, stream>>>(tmpf, xf, F(SI.ln1g) + i * kD,
                                                 F(SI.ln1b) + i * kD, seq, 0, xf, xbf, Mr);
      gemm(xbf, W.c1T + (long)i * 9 * kDFF * kD, F(SI.c1b) + i * kDFF,
           nullptr, h1bf, Mr, kDFF, kD, T, 9, 4, 1);
      gemm(h1bf, W.c2T + (long)i * kD * kDFF, F(SI.c2b) + i * kD,
           tmpf, nullptr, Mr, kD, kDFF, T, 1, 0, 0);
      ln_kernel<<<cdiv(Mr, 8), 256, 0, stream>>>(tmpf, xf, F(SI.ln2g) + i * kD,
                                                 F(SI.ln2b) + i * kD, seq, 0, xf, xbf, Mr);
    }
  };

  // ------------------------------ encoder ------------------------------
  embed_enc<<<cdiv((long)kB * kTT * kD, 256), 256, 0, stream>>>(
      src_seq, src_pos, F(WORD_EMB), F(ENC_POS_EMB), xef, xebf);
  run_stack(SW[0], ENC, xef, xebf, kTT, src_seq);

  // --------------------------- duration head ---------------------------
  const int Me = kB * kTT;
  gemm(xebf, dpc1T, F(DP_C1B), tmpf, nullptr, Me, kDPF, kD, kTT, 3, 1, 0);
  ln_kernel<<<cdiv(Me, 8), 256, 0, stream>>>(tmpf, nullptr, F(DP_LN1G), F(DP_LN1B),
                                             nullptr, 1, dpaf, dpab, Me);
  gemm(dpab, dpc2T, F(DP_C2B), tmpf, nullptr, Me, kDPF, kDPF, kTT, 3, 1, 0);
  ln_kernel<<<cdiv(Me, 8), 256, 0, stream>>>(tmpf, nullptr, F(DP_LN2G), F(DP_LN2B),
                                             nullptr, 1, dpbf, nullptr, Me);
  durdot<<<cdiv(Me, 8), 256, 0, stream>>>(dpbf, F(DP_LINW), F(DP_LINB), out + kMEL_TOTAL);

  // -------------------------- length regulate --------------------------
  cumsum_k<<<1, 32, 0, stream>>>(len_tgt, cum);
  lengthreg<<<cdiv((long)kB * kTM, 8), 256, 0, stream>>>(xef, cum, mel_pos,
                                                         F(DEC_POS_EMB), xdf, xdbf);

  // ------------------------------ decoder ------------------------------
  run_stack(SW[1], DEC, xdf, xdbf, kTM, mel_pos);

  // final x already multiplied by (mel_pos != 0) == keep mask
  gemm(xdbf, melT, F(MEL_B), out, nullptr, kB * kTM, kNM, kD, kTM, 1, 0, 0);
}
